// LEFTNetZ_73306501808914
// MI455X (gfx1250) — compile-verified, matrix operand order checked by reading, not run on hardware
//
#include <hip/hip_runtime.h>
#include <math.h>

// ---------------------------------------------------------------------------
// Types for CDNA5 WMMA (wave32, 16x16x32 bf16 -> f32)
// ---------------------------------------------------------------------------
typedef __attribute__((ext_vector_type(16))) __bf16 v16bf;
typedef __attribute__((ext_vector_type(8)))  __bf16 v8bf;
typedef __attribute__((ext_vector_type(8)))  float  v8f;

#define HID 128
#define PI_F 3.14159265358979f

__device__ __forceinline__ float silu_f(float x) { return x / (1.f + __expf(-x)); }

// ---------------------------------------------------------------------------
// Generic tiled GEMM: Y[M x N] = act(X[M x K] @ W[K x N] + bias)
// X fp32; weights pre-transposed to bf16 Wt[N][K] so B fragments are two
// contiguous global_load_b128 per fragment (L2/L0 resident).
// Block = 128 threads = 4 waves; block tile = 128(M) x 64(N).
// Each wave: 32 rows via 2 A-fragments x 4 B-fragments = 8 WMMAs / K-step.
// Requires: K % 32 == 0, N % 64 == 0 (true for all GEMMs in this model).
// ---------------------------------------------------------------------------
template <int ACT>  // 0 = none, 1 = silu
__global__ void __launch_bounds__(128)
k_gemm(const float* __restrict__ X, const __bf16* __restrict__ Wt,
       const float* __restrict__ bias, float* __restrict__ Y,
       int M, int K, int N) {
  __shared__ __bf16 As[128][40];  // 128 rows x 32 K, stride 40 keeps 16B align

  const int tid  = threadIdx.x;
  const int wv   = tid >> 5;
  const int lane = tid & 31;
  const int half = lane >> 4;
  const int lm   = lane & 15;
  const int row0 = blockIdx.x * 128;
  const int col0 = blockIdx.y * 64;

  v8f acc[2][4] = {};

  for (int kk = 0; kk < K; kk += 32) {
    // --- stage A tile (fp32 -> bf16), 128x32, 32 elems/thread ---
#pragma unroll
    for (int rr = 0; rr < 2; ++rr) {
      int r  = (tid >> 1) + rr * 64;
      int cb = (tid & 1) * 16;
      int gr = row0 + r;
      if (gr < M) {
        const float* src = X + (size_t)gr * K + kk + cb;
#pragma unroll
        for (int c = 0; c < 16; ++c) As[r][cb + c] = (__bf16)src[c];
      } else {
#pragma unroll
        for (int c = 0; c < 16; ++c) As[r][cb + c] = (__bf16)0.f;
      }
    }
    __syncthreads();

    // A fragments: lane lm = row-in-16, halves pick K {0..7,16..23}/{8..15,24..31}
    v16bf af[2];
#pragma unroll
    for (int f = 0; f < 2; ++f) {
      const __bf16* arow = &As[wv * 32 + f * 16 + lm][0];
      v8bf alo = *(const v8bf*)(arow + half * 8);
      v8bf ahi = *(const v8bf*)(arow + 16 + half * 8);
      af[f] = __builtin_shufflevector(alo, ahi, 0, 1, 2, 3, 4, 5, 6, 7,
                                      8, 9, 10, 11, 12, 13, 14, 15);
    }
#pragma unroll
    for (int g = 0; g < 4; ++g) {
      // B fragment straight from global (weights hot in L0/L2)
      const __bf16* wrow = Wt + (size_t)(col0 + g * 16 + lm) * K + kk;
      v8bf blo = *(const v8bf*)(wrow + half * 8);
      v8bf bhi = *(const v8bf*)(wrow + 16 + half * 8);
      v16bf b = __builtin_shufflevector(blo, bhi, 0, 1, 2, 3, 4, 5, 6, 7,
                                        8, 9, 10, 11, 12, 13, 14, 15);
#pragma unroll
      for (int f = 0; f < 2; ++f)
        acc[f][g] = __builtin_amdgcn_wmma_f32_16x16x32_bf16(
            /*neg_a=*/false, af[f], /*neg_b=*/false, b,
            /*c_mod=*/(short)0, acc[f][g], /*reuse_a=*/false, /*reuse_b=*/false);
    }
    __syncthreads();
  }

  // epilogue: C layout -> row = f*16 + half*8 + r, col = g*16 + lm
#pragma unroll
  for (int g = 0; g < 4; ++g) {
    int col = col0 + g * 16 + lm;
    float bv = bias ? bias[col] : 0.f;
#pragma unroll
    for (int f = 0; f < 2; ++f) {
#pragma unroll
      for (int r = 0; r < 8; ++r) {
        int grow = row0 + wv * 32 + f * 16 + half * 8 + r;
        if (grow < M) {
          float v = acc[f][g][r] + bv;
          if (ACT == 1) v = silu_f(v);
          Y[(size_t)grow * N + col] = v;
        }
      }
    }
  }
}

// ---------------------------------------------------------------------------
// Utility kernels
// ---------------------------------------------------------------------------
// dst[n*K + k] = (bf16) src[k*N + n]   (weight transpose, once per call)
__global__ void k_convert_transpose_bf16(const float* __restrict__ src,
                                         __bf16* __restrict__ dst,
                                         int K, int N) {
  size_t total = (size_t)K * N;
  for (size_t i = blockIdx.x * (size_t)blockDim.x + threadIdx.x; i < total;
       i += (size_t)gridDim.x * blockDim.x) {
    int n = (int)(i / K);
    int k = (int)(i % K);
    dst[i] = (__bf16)src[(size_t)k * N + n];
  }
}

__global__ void k_zero(float* p, size_t n) {
  for (size_t i = blockIdx.x * (size_t)blockDim.x + threadIdx.x; i < n;
       i += (size_t)gridDim.x * blockDim.x)
    p[i] = 0.f;
}

__global__ void k_mul_inplace(float* __restrict__ a, const float* __restrict__ b,
                              size_t n) {
  for (size_t i = blockIdx.x * (size_t)blockDim.x + threadIdx.x; i < n;
       i += (size_t)gridDim.x * blockDim.x)
    a[i] *= b[i];
}

__global__ void k_scale_rows(float* __restrict__ y, const float* __restrict__ s,
                             size_t rows, int cols) {
  size_t n = rows * (size_t)cols;
  for (size_t i = blockIdx.x * (size_t)blockDim.x + threadIdx.x; i < n;
       i += (size_t)gridDim.x * blockDim.x)
    y[i] *= s[i / cols];
}

// ---------------------------------------------------------------------------
// Node embedding: h0 = LayerNorm(ne_w[z] + ne_b); also seeds s = h0.
// ---------------------------------------------------------------------------
__global__ void __launch_bounds__(128)
k_node_embed(const int* __restrict__ z, const float* __restrict__ ne_w,
             const float* __restrict__ ne_b, float* __restrict__ h0,
             float* __restrict__ s) {
  __shared__ float red[128];
  int n = blockIdx.x, c = threadIdx.x;
  float x = ne_w[(size_t)z[n] * HID + c] + ne_b[c];
  red[c] = x;
  __syncthreads();
  for (int st = 64; st > 0; st >>= 1) { if (c < st) red[c] += red[c + st]; __syncthreads(); }
  float mean = red[0] * (1.f / HID);
  __syncthreads();
  float d = x - mean;
  red[c] = d * d;
  __syncthreads();
  for (int st = 64; st > 0; st >>= 1) { if (c < st) red[c] += red[c + st]; __syncthreads(); }
  float var = red[0] * (1.f / HID);
  float y = d * rsqrtf(var + 1e-5f);
  h0[(size_t)n * HID + c] = y;
  s[(size_t)n * HID + c] = y;
}

// xln = LayerNorm(s) * gamma + beta
__global__ void __launch_bounds__(128)
k_ln_affine(const float* __restrict__ X, const float* __restrict__ gamma,
            const float* __restrict__ beta, float* __restrict__ Y) {
  __shared__ float red[128];
  int n = blockIdx.x, c = threadIdx.x;
  float x = X[(size_t)n * HID + c];
  red[c] = x;
  __syncthreads();
  for (int st = 64; st > 0; st >>= 1) { if (c < st) red[c] += red[c + st]; __syncthreads(); }
  float mean = red[0] * (1.f / HID);
  __syncthreads();
  float d = x - mean;
  red[c] = d * d;
  __syncthreads();
  for (int st = 64; st > 0; st >>= 1) { if (c < st) red[c] += red[c + st]; __syncthreads(); }
  float var = red[0] * (1.f / HID);
  Y[(size_t)n * HID + c] = d * rsqrtf(var + 1e-5f) * gamma[c] + beta[c];
}

// ---------------------------------------------------------------------------
// Edge geometry + RBF. One wave per edge (4 edges / 128-thread block).
// frame[e][d][f], f: 0=ed, 1=ec, 2=ev
// ---------------------------------------------------------------------------
__global__ void __launch_bounds__(128)
k_edge_geom(const float* __restrict__ pos, const int* __restrict__ ei, int E,
            float* __restrict__ r_emb, float* __restrict__ rbounds,
            float* __restrict__ frame) {
  int e = blockIdx.x * 4 + (threadIdx.x >> 5);
  int lane = threadIdx.x & 31;
  if (e >= E) return;
  int j = ei[e], i = ei[E + e];
  float pjx = pos[3 * j], pjy = pos[3 * j + 1], pjz = pos[3 * j + 2];
  float pix = pos[3 * i], piy = pos[3 * i + 1], piz = pos[3 * i + 2];
  float vx = pjx - pix, vy = pjy - piy, vz = pjz - piz;
  float dist = sqrtf(vx * vx + vy * vy + vz * vz);
  float idn = 1.f / (dist + 1e-10f);
  float edx = vx * idn, edy = vy * idn, edz = vz * idn;
  float cx = piy * pjz - piz * pjy;
  float cy = piz * pjx - pix * pjz;
  float cz = pix * pjy - piy * pjx;
  float icn = 1.f / (sqrtf(cx * cx + cy * cy + cz * cz) + 1e-10f);
  float ecx = cx * icn, ecy = cy * icn, ecz = cz * icn;
  float evx = edy * ecz - edz * ecy;
  float evy = edz * ecx - edx * ecz;
  float evz = edx * ecy - edy * ecx;
  float rb = 0.5f * (__cosf(dist * (PI_F / 6.f)) + 1.f);
  if (lane == 0) {
    float* fr = frame + (size_t)e * 9;
    fr[0] = edx; fr[1] = ecx; fr[2] = evx;
    fr[3] = edy; fr[4] = ecy; fr[5] = evy;
    fr[6] = edz; fr[7] = ecz; fr[8] = evz;
    rbounds[e] = rb;
  }
  const float start = 0.00247875218f;  // exp(-6)
  float tmp = (2.f / 96.f) * (1.f - start);
  float beta = 1.f / (tmp * tmp);
  float env = rb * (dist < 6.f ? 1.f : 0.f);
  float expd = __expf(-dist);
  for (int k = lane; k < 96; k += 32) {
    float mean = start + (1.f - start) * (float)k * (1.f / 95.f);
    float t = expd - mean;
    r_emb[(size_t)e * 96 + k] = env * __expf(-beta * t * t);
  }
}

// ---------------------------------------------------------------------------
// Edge scatters (segment sums via float atomics). One edge per 128-thr block.
// ---------------------------------------------------------------------------
__global__ void __launch_bounds__(128)
k_scatter_s(const int* __restrict__ ei, int E, const float* __restrict__ h0,
            const float* __restrict__ r_hid, float* __restrict__ s) {
  int e = blockIdx.x, h = threadIdx.x;
  int j = ei[e], i = ei[E + e];
  atomicAdd(&s[(size_t)i * HID + h],
            h0[(size_t)j * HID + h] * r_hid[(size_t)e * HID + h]);
}

__global__ void __launch_bounds__(128)
k_scatter_S(const int* __restrict__ ei, int E, const float* __restrict__ s_lin,
            const float* __restrict__ r_hid, const float* __restrict__ frame,
            float* __restrict__ S) {
  int e = blockIdx.x, h = threadIdx.x;
  int j = ei[e], i = ei[E + e];
  float coef = s_lin[(size_t)j * HID + h] * r_hid[(size_t)e * HID + h];
  const float* fr = frame + (size_t)e * 9;
#pragma unroll
  for (int d = 0; d < 3; ++d)
    atomicAdd(&S[((size_t)i * 3 + d) * HID + h], coef * fr[d * 3 + 0]);  // ed
}

// m = xh[j]*rbfh ; s += seg(m1) ; vec += seg(m3*ed/sqrt(HID))  (vec[j]==0 here)
__global__ void __launch_bounds__(128)
k_scatter_m(const int* __restrict__ ei, int E, const float* __restrict__ xh,
            const float* __restrict__ rbfh, const float* __restrict__ frame,
            float* __restrict__ s, float* __restrict__ vec) {
  int e = blockIdx.x, h = threadIdx.x;
  int j = ei[e], i = ei[E + e];
  const float* xr = xh + (size_t)j * 384;
  const float* rr = rbfh + (size_t)e * 384;
  atomicAdd(&s[(size_t)i * HID + h], xr[h] * rr[h]);
  float m3 = xr[256 + h] * rr[256 + h] * 0.08838834764f;  // 1/sqrt(128)
  const float* fr = frame + (size_t)e * 9;
#pragma unroll
  for (int d = 0; d < 3; ++d)
    atomicAdd(&vec[((size_t)i * 3 + d) * HID + h], m3 * fr[d * 3 + 0]);
}

// ---------------------------------------------------------------------------
// scalarize for S[i] and S[j] -> ew[:, 0:256] (scaled by rbounds)
// tiny 3->32->1 MLP stays on VALU (not WMMA-shaped)
// ---------------------------------------------------------------------------
__global__ void __launch_bounds__(128)
k_scalarize(const int* __restrict__ ei, int E, const float* __restrict__ S,
            const float* __restrict__ frame, const float* __restrict__ rbounds,
            const float* __restrict__ w1, const float* __restrict__ b1,
            const float* __restrict__ w2, const float* __restrict__ b2,
            float* __restrict__ ew) {
  __shared__ float sw1[96], sb1[32], sw2[32], sfr[9];
  __shared__ float sb2, srb;
  int e = blockIdx.x, h = threadIdx.x;
  if (h < 96) sw1[h] = w1[h];
  if (h < 32) { sb1[h] = b1[h]; sw2[h] = w2[h]; }
  if (h < 9) sfr[h] = frame[(size_t)e * 9 + h];
  if (h == 0) { sb2 = b2[0]; srb = rbounds[e]; }
  __syncthreads();
  int node_i = ei[E + e];  // -> scalar3
  int node_j = ei[e];      // -> scalar4
#pragma unroll
  for (int sel = 0; sel < 2; ++sel) {
    int n = sel ? node_j : node_i;
    float s0 = S[((size_t)n * 3 + 0) * HID + h];
    float s1 = S[((size_t)n * 3 + 1) * HID + h];
    float s2 = S[((size_t)n * 3 + 2) * HID + h];
    float sc0 = s0 * sfr[0] + s1 * sfr[3] + s2 * sfr[6];
    float sc1 = fabsf(s0 * sfr[1] + s1 * sfr[4] + s2 * sfr[7]);
    float sc2 = s0 * sfr[2] + s1 * sfr[5] + s2 * sfr[8];
    float acc = sb2 + sc0;
#pragma unroll
    for (int u = 0; u < 32; ++u) {
      float hu = sc0 * sw1[u] + sc1 * sw1[32 + u] + sc2 * sw1[64 + u] + sb1[u];
      acc += silu_f(hu) * sw2[u];
    }
    ew[(size_t)e * 480 + sel * 128 + h] = acc * srb;
  }
}

// ew[:,256:384] = r_hid ; ew[:,384:480] = r_emb
__global__ void k_pack_ew(const float* __restrict__ r_hid,
                          const float* __restrict__ r_emb,
                          float* __restrict__ ew, size_t E) {
  size_t n = E * 224;
  for (size_t idx = blockIdx.x * (size_t)blockDim.x + threadIdx.x; idx < n;
       idx += (size_t)gridDim.x * blockDim.x) {
    size_t e = idx / 224;
    int c = (int)(idx % 224);
    if (c < 128) ew[e * 480 + 256 + c] = r_hid[e * HID + c];
    else         ew[e * 480 + 384 + (c - 128)] = r_emb[e * 96 + (c - 128)];
  }
}

// ---------------------------------------------------------------------------
// FTE: per-node invariants from vp = vec @ fte_eq_w
// ---------------------------------------------------------------------------
__global__ void __launch_bounds__(128)
k_fte_node(const float* __restrict__ vp, const float* __restrict__ s,
           float* __restrict__ cat, float* __restrict__ vdot) {
  int n = blockIdx.x, h = threadIdx.x;
  float ss = 0.f, dt = 0.f;
#pragma unroll
  for (int d = 0; d < 3; ++d) {
    float v1 = vp[((size_t)n * 3 + d) * 256 + h];
    float v2 = vp[((size_t)n * 3 + d) * 256 + 128 + h];
    ss += v1 * v1;
    dt += v1 * v2;
  }
  cat[(size_t)n * 256 + h] = s[(size_t)n * HID + h];
  cat[(size_t)n * 256 + 128 + h] = sqrtf(ss + 1e-10f);
  vdot[(size_t)n * HID + h] = dt * 0.08838834764f;  // / sqrt(128)
}

// s_final -> out = s@out_w; accumulate per-graph sums
__global__ void __launch_bounds__(128)
k_final_node(const float* __restrict__ s, const float* __restrict__ xh2,
             const float* __restrict__ vdot, const float* __restrict__ out_w,
             const int* __restrict__ batch, float* __restrict__ acc) {
  __shared__ float red[128];
  int n = blockIdx.x, h = threadIdx.x;
  float a1 = xh2[(size_t)n * 384 + h];
  float a2 = xh2[(size_t)n * 384 + 128 + h];
  float sn = s[(size_t)n * HID + h] +
             (a1 + a2 * vdot[(size_t)n * HID + h]) * 0.70710678118f;
  red[h] = sn * out_w[h];
  __syncthreads();
  for (int st = 64; st > 0; st >>= 1) { if (h < st) red[h] += red[h + st]; __syncthreads(); }
  if (h == 0) atomicAdd(&acc[batch[n]], red[0]);
}

__global__ void k_finalize(const float* __restrict__ acc,
                           const float* __restrict__ out_b,
                           float* __restrict__ out, int ngraph, float inv_cnt) {
  int g = threadIdx.x;
  if (g < ngraph) out[g] = acc[g] * inv_cnt + out_b[0];
}

// ---------------------------------------------------------------------------
// Host-side orchestration
// ---------------------------------------------------------------------------
static inline int gs_blocks(size_t n) {
  size_t b = (n + 255) / 256;
  return (int)(b < 4096 ? (b ? b : 1) : 4096);
}

extern "C" void kernel_launch(void* const* d_in, const int* in_sizes, int n_in,
                              void* d_out, int out_size, void* d_ws, size_t ws_size,
                              hipStream_t stream) {
  (void)n_in; (void)ws_size;
  const float* pos    = (const float*)d_in[0];
  const int*   z      = (const int*)d_in[1];
  const int*   batch  = (const int*)d_in[2];
  const int*   ei     = (const int*)d_in[3];
  const float* rl_w1  = (const float*)d_in[4];
  const float* rl_b1  = (const float*)d_in[5];
  const float* rl_w2  = (const float*)d_in[6];
  const float* rl_b2  = (const float*)d_in[7];
  const float* ne_w   = (const float*)d_in[8];
  const float* ne_b   = (const float*)d_in[9];
  const float* sv_w   = (const float*)d_in[10];
  const float* sv_b   = (const float*)d_in[11];
  const float* l3_w1  = (const float*)d_in[12];
  const float* l3_b1  = (const float*)d_in[13];
  const float* l3_w2  = (const float*)d_in[14];
  const float* l3_b2  = (const float*)d_in[15];
  const float* mpln_w = (const float*)d_in[16];
  const float* mpln_b = (const float*)d_in[17];
  const float* xp_w1  = (const float*)d_in[18];
  const float* xp_b1  = (const float*)d_in[19];
  const float* xp_w2  = (const float*)d_in[20];
  const float* xp_b2  = (const float*)d_in[21];
  const float* rp_w   = (const float*)d_in[22];
  const float* rp_b   = (const float*)d_in[23];
  const float* ip_w1  = (const float*)d_in[24];
  const float* ip_b1  = (const float*)d_in[25];
  const float* ip_w2  = (const float*)d_in[26];
  const float* ip_b2  = (const float*)d_in[27];
  const float* fte_eq = (const float*)d_in[28];
  const float* fte_w1 = (const float*)d_in[29];
  const float* fte_b1 = (const float*)d_in[30];
  const float* fte_w2 = (const float*)d_in[31];
  const float* fte_b2 = (const float*)d_in[32];
  const float* outw   = (const float*)d_in[33];
  const float* outb   = (const float*)d_in[34];

  const int N = in_sizes[0] / 3;
  const int E = in_sizes[3] / 2;
  float* out = (float*)d_out;
  const int ngraph = out_size;            // 4
  const float inv_cnt = (float)ngraph / (float)N;

  // bump allocator over d_ws
  size_t off = 0;
  char* base = (char*)d_ws;
  auto alloc = [&](size_t bytes) -> void* {
    void* p = base + off;
    off = (off + bytes + 255) & ~(size_t)255;
    return p;
  };
  auto allocf = [&](size_t n) -> float* { return (float*)alloc(n * sizeof(float)); };
  auto allocb = [&](size_t n) -> __bf16* { return (__bf16*)alloc(n * sizeof(__bf16)); };

  // bf16 transposed weight copies (Wt[N][K])
  __bf16* b_rl_w1 = allocb(96 * 128);
  __bf16* b_rl_w2 = allocb(128 * 128);
  __bf16* b_sv_w  = allocb(128 * 128);
  __bf16* b_xp_w1 = allocb(128 * 128);
  __bf16* b_xp_w2 = allocb(128 * 384);
  __bf16* b_rp_w  = allocb(96 * 384);
  __bf16* b_ip_w1 = allocb(480 * 384);
  __bf16* b_ip_w2 = allocb(384 * 384);
  __bf16* b_fteeq = allocb(128 * 256);
  __bf16* b_ftew1 = allocb(256 * 128);
  __bf16* b_ftew2 = allocb(128 * 384);

  // node buffers
  float* h0    = allocf((size_t)N * 128);
  float* s     = allocf((size_t)N * 128);
  float* s_lin = allocf((size_t)N * 128);
  float* Sbuf  = allocf((size_t)N * 384);   // N x 3 x 128
  float* xln   = allocf((size_t)N * 128);
  float* xp_t  = allocf((size_t)N * 128);
  float* xh    = allocf((size_t)N * 384);
  float* vec   = allocf((size_t)N * 384);   // N x 3 x 128
  float* vp    = allocf((size_t)N * 768);   // (3N) x 256
  float* cat   = allocf((size_t)N * 256);
  float* vdot  = allocf((size_t)N * 128);
  float* f1    = allocf((size_t)N * 128);
  float* xh2   = allocf((size_t)N * 384);
  float* gacc  = allocf(16);

  // edge buffers
  float* r_emb   = allocf((size_t)E * 96);
  float* rbounds = allocf((size_t)E);
  float* frame   = allocf((size_t)E * 9);
  float* rh_tmp  = allocf((size_t)E * 128);
  float* r_hid   = allocf((size_t)E * 128);
  float* ew      = allocf((size_t)E * 480);
  float* t_rp    = allocf((size_t)E * 384);  // becomes rbfh in-place
  float* t_ip1   = allocf((size_t)E * 384);
  float* t_ip2   = allocf((size_t)E * 384);

  // --- weight convert + transpose (fp32 KxN -> bf16 NxK; tiny, L2-hot) ---
  struct { const float* s; __bf16* d; int K, Nn; } cvt[] = {
      {rl_w1, b_rl_w1, 96, 128},   {rl_w2, b_rl_w2, 128, 128},
      {sv_w, b_sv_w, 128, 128},    {xp_w1, b_xp_w1, 128, 128},
      {xp_w2, b_xp_w2, 128, 384},  {rp_w, b_rp_w, 96, 384},
      {ip_w1, b_ip_w1, 480, 384},  {ip_w2, b_ip_w2, 384, 384},
      {fte_eq, b_fteeq, 128, 256}, {fte_w1, b_ftew1, 256, 128},
      {fte_w2, b_ftew2, 128, 384}};
  for (auto& c : cvt)
    k_convert_transpose_bf16<<<gs_blocks((size_t)c.K * c.Nn), 256, 0, stream>>>(
        c.s, c.d, c.K, c.Nn);

  // --- zero accumulators ---
  k_zero<<<gs_blocks((size_t)N * 384), 256, 0, stream>>>(Sbuf, (size_t)N * 384);
  k_zero<<<gs_blocks((size_t)N * 384), 256, 0, stream>>>(vec, (size_t)N * 384);
  k_zero<<<1, 32, 0, stream>>>(gacc, 16);

  // --- node embedding: h0 (and s = h0) ---
  k_node_embed<<<N, 128, 0, stream>>>(z, ne_w, ne_b, h0, s);

  if (E > 0) {
    // --- edge geometry + RBF ---
    k_edge_geom<<<(E + 3) / 4, 128, 0, stream>>>(pos, ei, E, r_emb, rbounds, frame);

    // --- radial MLP: r_hid = (silu(r_emb@rl_w1+b)@rl_w2+b) * rbounds ---
    dim3 g1((E + 127) / 128, 128 / 64);
    k_gemm<1><<<g1, 128, 0, stream>>>(r_emb, b_rl_w1, rl_b1, rh_tmp, E, 96, 128);
    k_gemm<0><<<g1, 128, 0, stream>>>(rh_tmp, b_rl_w2, rl_b2, r_hid, E, 128, 128);
    k_scale_rows<<<gs_blocks((size_t)E * 128), 256, 0, stream>>>(r_hid, rbounds, E, 128);

    // --- s = h0 + seg_sum(h0[j]*r_hid) ---
    k_scatter_s<<<E, 128, 0, stream>>>(ei, E, h0, r_hid, s);
  }

  // --- s_lin = silu(s@sv_w+b) ---
  dim3 gn128((N + 127) / 128, 128 / 64);
  k_gemm<1><<<gn128, 128, 0, stream>>>(s, b_sv_w, sv_b, s_lin, N, 128, 128);

  if (E > 0) {
    // --- S = seg_sum(s_lin[j] * r_hid * ed) ---
    k_scatter_S<<<E, 128, 0, stream>>>(ei, E, s_lin, r_hid, frame, Sbuf);

    // --- scalarize(S[i]), scalarize(S[j]) -> ew[:,0:256]; pack rest ---
    k_scalarize<<<E, 128, 0, stream>>>(ei, E, Sbuf, frame, rbounds,
                                       l3_w1, l3_b1, l3_w2, l3_b2, ew);
    k_pack_ew<<<gs_blocks((size_t)E * 224), 256, 0, stream>>>(r_hid, r_emb, ew, (size_t)E);
  }

  // --- xh = silu(LN(s)*g+b @ xp_w1 + b) @ xp_w2 + b ---
  k_ln_affine<<<N, 128, 0, stream>>>(s, mpln_w, mpln_b, xln);
  k_gemm<1><<<gn128, 128, 0, stream>>>(xln, b_xp_w1, xp_b1, xp_t, N, 128, 128);
  dim3 gn384((N + 127) / 128, 384 / 64);
  k_gemm<0><<<gn384, 128, 0, stream>>>(xp_t, b_xp_w2, xp_b2, xh, N, 128, 384);

  if (E > 0) {
    // --- rbfh = (r_emb@rp_w+b) * (silu(ew@ip_w1+b)@ip_w2+b)  [the big GEMMs] ---
    dim3 ge384((E + 127) / 128, 384 / 64);
    k_gemm<0><<<ge384, 128, 0, stream>>>(r_emb, b_rp_w, rp_b, t_rp, E, 96, 384);
    k_gemm<1><<<ge384, 128, 0, stream>>>(ew, b_ip_w1, ip_b1, t_ip1, E, 480, 384);
    k_gemm<0><<<ge384, 128, 0, stream>>>(t_ip1, b_ip_w2, ip_b2, t_ip2, E, 384, 384);
    k_mul_inplace<<<gs_blocks((size_t)E * 384), 256, 0, stream>>>(t_rp, t_ip2,
                                                                  (size_t)E * 384);
    // --- message scatter: s += seg(m1); vec += seg(m3*ed/sqrt(HID)) ---
    k_scatter_m<<<E, 128, 0, stream>>>(ei, E, xh, t_rp, frame, s, vec);
  }

  // --- FTE: vp = vec @ fte_eq_w (no bias) ---
  dim3 g3n((3 * N + 127) / 128, 256 / 64);
  k_gemm<0><<<g3n, 128, 0, stream>>>(vec, b_fteeq, nullptr, vp, 3 * N, 128, 256);
  k_fte_node<<<N, 128, 0, stream>>>(vp, s, cat, vdot);

  // --- xh2 = silu(cat@fte_w1+b)@fte_w2+b ---
  k_gemm<1><<<gn128, 128, 0, stream>>>(cat, b_ftew1, fte_b1, f1, N, 256, 128);
  k_gemm<0><<<gn384, 128, 0, stream>>>(f1, b_ftew2, fte_b2, xh2, N, 128, 384);

  // --- final: s update, readout, graph mean ---
  k_final_node<<<N, 128, 0, stream>>>(s, xh2, vdot, outw, batch, gacc);
  k_finalize<<<1, 32, 0, stream>>>(gacc, outb, out, ngraph, inv_cnt);
}
